// GroupGMM_61426622267739
// MI455X (gfx1250) — matrix operand order repaired and checked by reading, not compiled, not run on hardware
//
#include <hip/hip_runtime.h>
#include <hip/hip_bf16.h>
#include <math.h>

typedef __attribute__((ext_vector_type(16))) __bf16 v16bf;
typedef __attribute__((ext_vector_type(8)))  __bf16 v8bf;
typedef __attribute__((ext_vector_type(8)))  float  v8f;
typedef __attribute__((ext_vector_type(4)))  unsigned int v4u;
typedef __attribute__((ext_vector_type(4)))  int v4i;

#define AS1 __attribute__((address_space(1)))
#define AS3 __attribute__((address_space(3)))

#define B_DIM   8192
#define I_DIM   512
#define G_DIM   32
#define C_DIM   16
#define CD_DIM  512
#define K_DIM   16384            // G*I
#define N_OUT   1040             // pi | mu | sigma == logits | loc | scale
#define KROW_BYTES 32768         // K_DIM * sizeof(bf16)
#define LPITCH  144              // LDS row pitch bytes (72 bf16) for a 64-wide K slab
#define TILEB   18432            // 128 * LPITCH

#if __has_builtin(__builtin_amdgcn_global_load_async_to_lds_b128) && \
    __has_builtin(__builtin_amdgcn_s_wait_asynccnt)
#define GG_ASYNC 1
#else
#define GG_ASYNC 0
#endif

// ---------------------------------------------------------------------------
// Prep 1: Zbf[b][k] = bf16( g[b, k>>9] * x[b, k&511] ),  [8192 x 16384] bf16
// ---------------------------------------------------------------------------
__global__ __launch_bounds__(256) void gg_build_z(
    const float* __restrict__ x, const float* __restrict__ g, __bf16* __restrict__ Z)
{
  int tid = blockIdx.x * 256 + threadIdx.x;
  int b   = tid >> 11;
  int k   = (tid & 2047) << 3;
  float s = g[b * G_DIM + (k >> 9)];
  const float* xp = x + (size_t)b * I_DIM + (k & 511);
  v8bf o;
#pragma unroll
  for (int u = 0; u < 8; ++u) o[u] = (__bf16)(xp[u] * s);
  *(v8bf*)(Z + (size_t)b * K_DIM + k) = o;
}

// ---------------------------------------------------------------------------
// Prep 2: WallT[n][k] (bf16, [1040 x 16384]) transposed concat of weights.
// ---------------------------------------------------------------------------
__global__ void gg_build_wt(const float* __restrict__ Wmu, const float* __restrict__ Wsig,
                            const float* __restrict__ Wpi, __bf16* __restrict__ Wt)
{
  __shared__ __bf16 tile[32][33];
  int gg = blockIdx.x, it = blockIdx.y, nt = blockIdx.z;
  int i0 = it * 32, n0 = nt * 32;
  int tx = threadIdx.x, ty = threadIdx.y;       // block (32, 8)
#pragma unroll
  for (int r = 0; r < 4; ++r) {
    int col = n0 + tx;
    int ri  = i0 + ty + 8 * r;
    float v = 0.f;
    if (col < N_OUT) {
      if (col < 16)       v = Wpi[((size_t)gg * I_DIM + ri) * C_DIM  + col];
      else if (col < 528) v = Wmu[((size_t)gg * I_DIM + ri) * CD_DIM + (col - 16)];
      else                v = Wsig[((size_t)gg * I_DIM + ri) * CD_DIM + (col - 528)];
    }
    tile[ty + 8 * r][tx] = (__bf16)v;
  }
  __syncthreads();
#pragma unroll
  for (int r = 0; r < 4; ++r) {
    int n = n0 + ty + 8 * r;
    if (n < N_OUT)
      Wt[(size_t)n * K_DIM + (size_t)gg * I_DIM + i0 + tx] = tile[tx][ty + 8 * r];
  }
}

// ---------------------------------------------------------------------------
// Prep 3: gbias[b][c] = sum_g g[b][g] * b_cat[g][c]   ([8192 x 1040] f32)
// ---------------------------------------------------------------------------
__global__ __launch_bounds__(256) void gg_build_gb(
    const float* __restrict__ g, const float* __restrict__ bmu,
    const float* __restrict__ bsig, const float* __restrict__ bpi,
    float* __restrict__ gb)
{
  int b = blockIdx.x;
  for (int c = threadIdx.x; c < N_OUT; c += 256) {
    float acc = 0.f;
#pragma unroll 4
    for (int gg = 0; gg < G_DIM; ++gg) {
      float bb = (c < 16)  ? bpi[gg * C_DIM + c]
               : (c < 528) ? bmu[gg * CD_DIM + (c - 16)]
                           : bsig[gg * CD_DIM + (c - 528)];
      acc += g[b * G_DIM + gg] * bb;
    }
    gb[(size_t)b * N_OUT + c] = acc;
  }
}

// ---------------------------------------------------------------------------
// Main GEMM: out = Zbf @ WallT^T (+bias, softplus on sigma cols)
// 128(M) x NF*16(N) per WG; 8 waves; K staged via LDS in 64-wide slabs.
// Async path: 3-buffer, depth-2 GLOBAL_LOAD_ASYNC_TO_LDS_B128 pipeline
//   (uniform 8 async ops/thread/stage so s_wait_asynccnt(8) == "all but the
//    newest stage retired").
// Fallback: register-staged global_load_b128 -> ds_store_b128.
// ---------------------------------------------------------------------------
template <int NF>
__global__ __launch_bounds__(256) void gg_gemm(
    const __bf16* __restrict__ Z, const __bf16* __restrict__ Wt,
    const float* __restrict__ gb, float* __restrict__ out)
{
#if GG_ASYNC
  __shared__ __bf16 lA[3][128 * 72];
  __shared__ __bf16 lB[3][128 * 72];
#else
  __shared__ __bf16 lA[1][128 * 72];
  __shared__ __bf16 lB[1][128 * 72];
#endif

  const int t = threadIdx.x;
  const int bBase = blockIdx.x * 128;
  const int nBase = (NF == 8) ? (int)blockIdx.y * 128 : 1024;

  // ---- copy mapping: thread owns 4 rows (stride 32), one 16B chunk each
  const int rowc = t >> 3;             // 0..31
  const int offc = (t & 7) * 16;       // byte offset within 128B K-slab row
  const unsigned long long gA =
      (unsigned long long)Z  + (size_t)(bBase + rowc) * KROW_BYTES + offc;
  const unsigned long long gB =
      (unsigned long long)Wt + (size_t)(nBase + rowc) * KROW_BYTES + offc;

  // ---- WMMA fragment addressing (wave32)
  const int w = t >> 5, lane = t & 31;
  const int fr = lane & 15;            // A row m / B col n / C,D col N
  const int hh = lane >> 4;
  const char* aF = (const char*)&lA[0][0] + (w * 16 + fr) * LPITCH + hh * 16;
  const char* bF = (const char*)&lB[0][0] + fr * LPITCH + hh * 16;

  v8f acc[NF] = {};

  auto computeSlab = [&](const char* aBuf, const char* bBuf) {
    v16bf av[2];
    v16bf bv[2][NF];
#pragma unroll
    for (int koi = 0; koi < 2; ++koi) {
      v8bf alo = *(const v8bf*)(aBuf + koi * 64);
      v8bf ahi = *(const v8bf*)(aBuf + koi * 64 + 32);
      av[koi]  = __builtin_shufflevector(alo, ahi,
                   0,1,2,3,4,5,6,7,8,9,10,11,12,13,14,15);
#pragma unroll
      for (int j = 0; j < NF; ++j) {
        v8bf blo = *(const v8bf*)(bBuf + j * (16 * LPITCH) + koi * 64);
        v8bf bhi = *(const v8bf*)(bBuf + j * (16 * LPITCH) + koi * 64 + 32);
        bv[koi][j] = __builtin_shufflevector(blo, bhi,
                       0,1,2,3,4,5,6,7,8,9,10,11,12,13,14,15);
      }
    }
#pragma unroll
    for (int koi = 0; koi < 2; ++koi)
#pragma unroll
      for (int j = 0; j < NF; ++j)
        acc[j] = __builtin_amdgcn_wmma_f32_16x16x32_bf16(
                   false, av[koi], false, bv[koi][j], (short)0, acc[j], false, false);
  };

#if GG_ASYNC
  // ------------- async DMA, 3 buffers, 2 stages in flight -------------
  const unsigned sAoff = (unsigned)(unsigned long long)(void*)&lA[0][0]
                         + rowc * LPITCH + offc;
  const unsigned sBoff = (unsigned)(unsigned long long)(void*)&lB[0][0]
                         + rowc * LPITCH + offc;

  // Exactly 8 async ops per thread per stage (uniform, so the asynccnt
  // threshold is a compile-time immediate). For NF==1 the extra B rows read
  // harmless in-workspace bytes and are never consumed from LDS.
  auto issue = [&](int s, int buf) {
    const unsigned ao = sAoff + buf * TILEB;
    const unsigned bo = sBoff + buf * TILEB;
#pragma unroll
    for (int j = 0; j < 4; ++j) {
      __builtin_amdgcn_global_load_async_to_lds_b128(
          (AS1 v4i*)(gA + (unsigned long long)j * 32 * KROW_BYTES
                        + (unsigned long long)s * 128),
          (AS3 v4i*)(ao + j * 32 * LPITCH), 0, 0);
      __builtin_amdgcn_global_load_async_to_lds_b128(
          (AS1 v4i*)(gB + (unsigned long long)j * 32 * KROW_BYTES
                        + (unsigned long long)s * 128),
          (AS3 v4i*)(bo + j * 32 * LPITCH), 0, 0);
    }
  };

  issue(0, 0);
  issue(1, 1);
  int buf = 0;                 // buffer holding stage s
  for (int s = 0; s < 256; ++s) {
    if (s < 255) __builtin_amdgcn_s_wait_asynccnt(8);  // stage s retired
    else         __builtin_amdgcn_s_wait_asynccnt(0);  // final stage retired
    __syncthreads();                                   // LDS buf valid for all
    if (s + 2 < 256) {
      const int nbuf = (buf + 2 >= 3) ? buf - 1 : buf + 2;
      issue(s + 2, nbuf);                              // DMA 2 slabs ahead
    }
    computeSlab(aF + buf * TILEB, bF + buf * TILEB);
    buf = (buf + 1 >= 3) ? 0 : buf + 1;
  }
#else
  // -------------------- register-staged fallback pipeline --------------------
  char* sA = (char*)&lA[0][0] + rowc * LPITCH + offc;
  char* sB = (char*)&lB[0][0] + rowc * LPITCH + offc;
  v4u ra[4];
  v4u rb[4] = {};
  auto loadRegs = [&](int s) {
#pragma unroll
    for (int j = 0; j < 4; ++j) {
      ra[j] = *(const v4u*)(gA + (size_t)j * 32 * KROW_BYTES + (size_t)s * 128);
      if (NF == 8 || (j == 0 && rowc < 16))
        rb[j] = *(const v4u*)(gB + (size_t)j * 32 * KROW_BYTES + (size_t)s * 128);
    }
  };

  loadRegs(0);
  for (int s = 0; s < 256; ++s) {
    __syncthreads();
#pragma unroll
    for (int j = 0; j < 4; ++j) {
      *(v4u*)(sA + j * 32 * LPITCH) = ra[j];
      *(v4u*)(sB + j * 32 * LPITCH) = rb[j];
    }
    __syncthreads();
    if (s < 255) loadRegs(s + 1);
    computeSlab(aF, bF);
  }
#endif

  // ---- epilogue: bias, softplus on sigma columns (col >= 528), store f32
#pragma unroll
  for (int j = 0; j < NF; ++j) {
    const int col = nBase + j * 16 + fr;
#pragma unroll
    for (int r = 0; r < 8; ++r) {
      const int bi = bBase + w * 16 + r + 8 * hh;   // C/D layout: M = r + 8*half
      float v = acc[j][r] + gb[(size_t)bi * N_OUT + col];
      if (col >= 528)
        v = fmaxf(v, 0.f) + log1pf(expf(-fabsf(v))) + 1e-7f;
      out[(size_t)bi * N_OUT + col] = v;
    }
  }
}

// ---------------------------------------------------------------------------
extern "C" void kernel_launch(void* const* d_in, const int* in_sizes, int n_in,
                              void* d_out, int out_size, void* d_ws, size_t ws_size,
                              hipStream_t stream)
{
  (void)in_sizes; (void)n_in; (void)out_size; (void)ws_size;
  const float* x    = (const float*)d_in[0];
  const float* g    = (const float*)d_in[1];
  const float* Wmu  = (const float*)d_in[2];
  const float* bmu  = (const float*)d_in[3];
  const float* Wsig = (const float*)d_in[4];
  const float* bsig = (const float*)d_in[5];
  const float* Wpi  = (const float*)d_in[6];
  const float* bpi  = (const float*)d_in[7];
  float* out = (float*)d_out;

  char* ws = (char*)d_ws;
  __bf16* Z    = (__bf16*)(ws);                                   // 268,435,456 B
  __bf16* Wt   = (__bf16*)(ws + (size_t)268435456);               //  34,078,720 B
  float*  gbuf = (float*) (ws + (size_t)268435456 + 34078720);    //  34,078,720 B

  gg_build_z <<<65536, 256, 0, stream>>>(x, g, Z);
  gg_build_wt<<<dim3(32, 16, 33), dim3(32, 8), 0, stream>>>(Wmu, Wsig, Wpi, Wt);
  gg_build_gb<<<8192, 256, 0, stream>>>(g, bmu, bsig, bpi, gbuf);

  gg_gemm<8><<<dim3(64, 8), 256, 0, stream>>>(Z, Wt, gbuf, out);  // cols 0..1023
  gg_gemm<1><<<dim3(64, 1), 256, 0, stream>>>(Z, Wt, gbuf, out);  // cols 1024..1039
}
